// SpatialAttentionModule_76347338653974
// MI455X (gfx1250) — matrix operand order repaired and checked by reading, not compiled, not run on hardware
//
#include <hip/hip_runtime.h>
#include <hip/hip_bf16.h>
#include <math.h>

typedef _Float16 v16h __attribute__((ext_vector_type(16)));
typedef _Float16 v8h  __attribute__((ext_vector_type(8)));
typedef float    v8f  __attribute__((ext_vector_type(8)));

#define BATCH 4
#define CCH   256
#define C4DIM 64
#define NPIX  4096   // 64*64

__device__ __forceinline__ int pack2h(float a, float b) {
    auto p = __builtin_amdgcn_cvt_pkrtz(a, b);      // v_cvt_pk_rtz_f16_f32
    return __builtin_bit_cast(int, p);
}

// A-operand fragment (16x32 f16, lane = M): elements 0..7 from p[0..7],
// elements 8..15 from p[16..23] (caller pre-offsets p by half*8).
__device__ __forceinline__ v16h load_a_frag(const _Float16* p) {
    v8h lo = *(const v8h*)(p);
    v8h hi = *(const v8h*)(p + 16);
    return __builtin_shufflevector(lo, hi,
        0, 1, 2, 3, 4, 5, 6, 7, 8, 9, 10, 11, 12, 13, 14, 15);
}

// ---------------- Kernel 1a: convert fused weight matrix to f16 ----------------
// Wh[o][c], o in 0..383 = [Wq(64) | Wk(64) | Wv(256)]
__global__ __launch_bounds__(256) void conv_w(
    const float* __restrict__ Wq, const float* __restrict__ Wk,
    const float* __restrict__ Wv, _Float16* __restrict__ Wh)
{
    const int idx = blockIdx.x * 256 + threadIdx.x;   // 0 .. 384*256-1
    const int o = idx >> 8;
    const int c = idx & 255;
    float v;
    if (o < 64)       v = Wq[(size_t)o * CCH + c];
    else if (o < 128) v = Wk[(size_t)(o - 64) * CCH + c];
    else              v = Wv[(size_t)(o - 128) * CCH + c];
    Wh[idx] = (_Float16)v;
}

// ---------------- Kernel 1b: transpose+convert x -> xt[b][n][c] f16 ----------------
__global__ __launch_bounds__(256) void transpose_x(
    const float* __restrict__ x, _Float16* __restrict__ xt)
{
    __shared__ _Float16 tile[64][66];                 // pad 66 -> conflict-free cols
    const int b  = blockIdx.z;
    const int c0 = blockIdx.y * 64;
    const int n0 = blockIdx.x * 64;
    const int tid = threadIdx.x;

    // load phase: pairs along n so DS writes are dword-wide
    {
        const int tx2 = tid & 31;       // n-pair index
        const int ty2 = tid >> 5;       // 0..7
        #pragma unroll
        for (int i = 0; i < 8; ++i) {
            const int c = ty2 + i * 8;
            const float* src = x + ((size_t)b * CCH + c0 + c) * NPIX + n0 + tx2 * 2;
            int pk = pack2h(src[0], src[1]);
            *(int*)&tile[c][tx2 * 2] = pk;
        }
    }
    __syncthreads();
    // store phase: rows of xt are contiguous in c
    {
        const int tx = tid & 63;        // c index
        const int ty = tid >> 6;        // 0..3
        #pragma unroll
        for (int i = 0; i < 16; ++i) {
            const int n = ty + i * 4;
            xt[((size_t)b * NPIX + n0 + n) * CCH + c0 + tx] = tile[tx][n];
        }
    }
}

// ---------------- Kernel 1c: WMMA QKV projection ----------------
// D[o,n] = Wh[o,:] * xt[:,n]; each wave: 16o x 32n tile, K=256 (8 chunks).
__global__ __launch_bounds__(256) void proj_wmma(
    const _Float16* __restrict__ Wh, const _Float16* __restrict__ xt,
    const float* __restrict__ bq, const float* __restrict__ bk,
    const float* __restrict__ bv,
    _Float16* __restrict__ Qh, _Float16* __restrict__ Kh,
    _Float16* __restrict__ Vh)
{
    const int tid  = threadIdx.x;
    const int lane = tid & 31;
    const int wv   = tid >> 5;
    const int hl   = (lane >> 4) & 1;
    const int l16  = lane & 15;
    const int b    = blockIdx.z;
    const int o0   = blockIdx.y * 16;                    // 0..368
    const int n0   = (blockIdx.x * 8 + wv) * 32;         // 0..4064

    const _Float16* wrow = Wh + (size_t)(o0 + l16) * CCH + hl * 8;
    const _Float16* x0 = xt + ((size_t)b * NPIX + n0 + l16) * CCH + hl * 8;
    const _Float16* x1 = x0 + 16 * CCH;

    v8f zero = {0.f, 0.f, 0.f, 0.f, 0.f, 0.f, 0.f, 0.f};
    v8f acc0 = zero, acc1 = zero;
    #pragma unroll
    for (int cc = 0; cc < CCH; cc += 32) {
        v16h aw = load_a_frag(wrow + cc);
        v16h b0 = load_a_frag(x0 + cc);
        v16h b1 = load_a_frag(x1 + cc);
        acc0 = __builtin_amdgcn_wmma_f32_16x16x32_f16(false, aw, false, b0, (short)0, acc0, false, false);
        acc1 = __builtin_amdgcn_wmma_f32_16x16x32_f16(false, aw, false, b1, (short)0, acc1, false, false);
    }

    // Epilogue: lane (l16,hl) holds o = o0 + 8*hl + r (r=0..7) at n = n0(+16) + l16.
    if (o0 < 128) {
        // Q/K: dest layout [n][o]; lane's 8 values are 8 contiguous o -> b128 store
        const float* bias = (o0 < 64) ? (bq + o0) : (bk + o0 - 64);
        const float s = (o0 < 64) ? 0.125f : 1.0f;
        _Float16* dst = (o0 < 64) ? Qh : Kh;
        const int oo = (o0 < 64) ? o0 : (o0 - 64);
        union { int i[4]; v8h v; } u0, u1;
        #pragma unroll
        for (int j = 0; j < 4; ++j) {
            const float ba = bias[8 * hl + 2 * j];
            const float bb = bias[8 * hl + 2 * j + 1];
            u0.i[j] = pack2h((acc0[2 * j] + ba) * s, (acc0[2 * j + 1] + bb) * s);
            u1.i[j] = pack2h((acc1[2 * j] + ba) * s, (acc1[2 * j + 1] + bb) * s);
        }
        *(v8h*)(dst + ((size_t)b * NPIX + n0 + l16) * C4DIM + oo + 8 * hl) = u0.v;
        *(v8h*)(dst + ((size_t)b * NPIX + n0 + 16 + l16) * C4DIM + oo + 8 * hl) = u1.v;
    } else {
        // V: dest layout [o][n]; 16 lanes of a row are contiguous f16
        const float* bias = bv + (o0 - 128);
        #pragma unroll
        for (int r = 0; r < 8; ++r) {
            const int o = (o0 - 128) + 8 * hl + r;
            const float bi = bias[8 * hl + r];
            _Float16* vr = Vh + ((size_t)b * CCH + o) * NPIX;
            vr[n0 + l16]      = (_Float16)(acc0[r] + bi);
            vr[n0 + 16 + l16] = (_Float16)(acc1[r] + bi);
        }
    }
}

// ---------------- Kernel 2: fused streaming attention (unchanged) ----------------
__global__ __launch_bounds__(256) void attn_fused(
    const float* __restrict__ x,
    const _Float16* __restrict__ Qh,
    const _Float16* __restrict__ Kh,
    const _Float16* __restrict__ Vh,
    const float* __restrict__ gamma,
    float* __restrict__ out)
{
    const int tid  = threadIdx.x;
    const int lane = tid & 31;
    const int wv   = tid >> 5;          // wave id 0..7
    const int hl   = (lane >> 4) & 1;   // half-wave
    const int l16  = lane & 15;
    const int b    = blockIdx.y;
    const int q    = blockIdx.x * 128 + wv * 16 + l16;

    const _Float16* qrow = Qh + ((size_t)b * NPIX + q) * C4DIM;
    v16h qb0 = load_a_frag(qrow + hl * 8);            // dims 0..31
    v16h qb1 = load_a_frag(qrow + 32 + hl * 8);       // dims 32..63

    v8f zero = {0.f, 0.f, 0.f, 0.f, 0.f, 0.f, 0.f, 0.f};
    v8f oacc[16];
    #pragma unroll
    for (int i = 0; i < 16; ++i) oacc[i] = zero;
    float rowsum = 0.f;

    for (int m = 0; m < NPIX; m += 32) {
        v8f s0 = zero, s1 = zero;
        {
            const _Float16* krow0 = Kh + ((size_t)b * NPIX + m + l16) * C4DIM;
            const _Float16* krow1 = krow0 + 16 * C4DIM;
            v16h a00 = load_a_frag(krow0 + hl * 8);
            v16h a01 = load_a_frag(krow0 + 32 + hl * 8);
            v16h a10 = load_a_frag(krow1 + hl * 8);
            v16h a11 = load_a_frag(krow1 + 32 + hl * 8);
            s0 = __builtin_amdgcn_wmma_f32_16x16x32_f16(false, a00, false, qb0, (short)0, s0, false, false);
            s0 = __builtin_amdgcn_wmma_f32_16x16x32_f16(false, a01, false, qb1, (short)0, s0, false, false);
            s1 = __builtin_amdgcn_wmma_f32_16x16x32_f16(false, a10, false, qb0, (short)0, s1, false, false);
            s1 = __builtin_amdgcn_wmma_f32_16x16x32_f16(false, a11, false, qb1, (short)0, s1, false, false);
        }

        float e0[8], e1[8];
        #pragma unroll
        for (int r = 0; r < 8; ++r) {
            float v0 = fminf(5.f, fmaxf(-5.f, s0[r]));
            float v1 = fminf(5.f, fmaxf(-5.f, s1[r]));
            e0[r] = __expf(v0);
            e1[r] = __expf(v1);
            rowsum += e0[r] + e1[r];
        }

        union { int i[8]; v16h v; } pb;
        #pragma unroll
        for (int r = 0; r < 4; ++r) {
            int lo = pack2h(e0[2 * r], e0[2 * r + 1]);
            int hi = pack2h(e1[2 * r], e1[2 * r + 1]);
            int lo_sw = __shfl_xor(lo, 16, 32);
            int hi_sw = __shfl_xor(hi, 16, 32);
            pb.i[r]     = lo;
            pb.i[r + 4] = hi;
        }

        const _Float16* vbase = Vh + ((size_t)b * CCH + l16) * NPIX + m + hl * 8;
        #pragma unroll
        for (int ct = 0; ct < 16; ++ct) {
            v16h va = load_a_frag(vbase + (size_t)(ct * 16) * NPIX);
            oacc[ct] = __builtin_amdgcn_wmma_f32_16x16x32_f16(false, va, false, pb.v, (short)0, oacc[ct], false, false);
        }
    }

    const float inv = 1.f / (rowsum + __shfl_xor(rowsum, 16, 32));
    const float g = gamma[0];
    #pragma unroll
    for (int ct = 0; ct < 16; ++ct) {
        const int c0 = ct * 16 + hl * 8;
        #pragma unroll
        for (int r = 0; r < 8; ++r) {
            const size_t idx = ((size_t)b * CCH + (c0 + r)) * NPIX + q;
            out[idx] = fmaf(g, oacc[ct][r] * inv, x[idx]);
        }
    }
}

extern "C" void kernel_launch(void* const* d_in, const int* in_sizes, int n_in,
                              void* d_out, int out_size, void* d_ws, size_t ws_size,
                              hipStream_t stream) {
    const float* x     = (const float*)d_in[0];
    const float* Wq    = (const float*)d_in[1];
    const float* bq    = (const float*)d_in[2];
    const float* Wk    = (const float*)d_in[3];
    const float* bk    = (const float*)d_in[4];
    const float* Wv    = (const float*)d_in[5];
    const float* bv    = (const float*)d_in[6];
    const float* gamma = (const float*)d_in[7];
    float* out = (float*)d_out;

    // Workspace (f16): Qh 2MB | Kh 2MB | Vh 8MB | xt 8MB | Wh 192KB  (~20.2MB)
    _Float16* Qh = (_Float16*)d_ws;
    _Float16* Kh = Qh + (size_t)BATCH * NPIX * C4DIM;
    _Float16* Vh = Kh + (size_t)BATCH * NPIX * C4DIM;
    _Float16* xt = Vh + (size_t)BATCH * CCH * NPIX;
    _Float16* Wh = xt + (size_t)BATCH * NPIX * CCH;

    conv_w<<<dim3(384), 256, 0, stream>>>(Wq, Wk, Wv, Wh);
    transpose_x<<<dim3(NPIX / 64, CCH / 64, BATCH), 256, 0, stream>>>(x, xt);
    proj_wmma<<<dim3(16, 384 / 16, BATCH), 256, 0, stream>>>(
        Wh, xt, bq, bk, bv, Qh, Kh, Vh);
    attn_fused<<<dim3(NPIX / 128, BATCH), 256, 0, stream>>>(
        x, Qh, Kh, Vh, gamma, out);
}
